// OptimizedClassificationHead_40785009443177
// MI455X (gfx1250) — matrix operand order repaired and checked by reading, not compiled, not run on hardware
//
#include <hip/hip_runtime.h>
#include <hip/hip_bf16.h>

// ---------------------------------------------------------------------------
// CDNA5 (gfx1250) top-M attention classification head.
// One workgroup (256 threads = 8 wave32) handles one of the 1920 sequences.
// All GEMMs run on v_wmma_f32_16x16x32_f16; all activations live in LDS.
// Round 5: sched_barrier-enforced half-tile software pipeline in the hot
// GEMM loops -> partial s_wait_dscnt, WMMA blocks overlap next tile's loads.
// ---------------------------------------------------------------------------

typedef _Float16 v8h  __attribute__((ext_vector_type(8)));
typedef _Float16 v16h __attribute__((ext_vector_type(16)));
typedef float    v8f  __attribute__((ext_vector_type(8)));
typedef int      v4i  __attribute__((vector_size(16)));

#define WMMA_F16(a, b, c) \
    __builtin_amdgcn_wmma_f32_16x16x32_f16(false, (a), false, (b), (short)0, (c), false, false)

#if defined(__AMDGCN__) && __has_builtin(__builtin_amdgcn_global_load_async_to_lds_b128) && \
    __has_builtin(__builtin_amdgcn_s_wait_asynccnt)
#define USE_ASYNC_FEAT 1
#define AS1_V4I(p) ((__attribute__((address_space(1))) v4i*)(unsigned long long)(p))
#define AS3_V4I(p) ((__attribute__((address_space(3))) v4i*)(unsigned long long)(p))
#endif

#if defined(__AMDGCN__) && __has_builtin(__builtin_amdgcn_sched_barrier)
#define SCHED_FENCE() __builtin_amdgcn_sched_barrier(0)
#else
#define SCHED_FENCE()
#endif

// A fragment (16x32 f16): lane holds row M=lane&15; halves = 16 contiguous K
// starting at hsel*16  -> two contiguous 16B LDS reads.
__device__ __forceinline__ v16h frag_a(const _Float16* p) {
    v8h lo = *(const v8h*)p;
    v8h hi = *(const v8h*)(p + 8);
    return __builtin_shufflevector(lo, hi, 0,1,2,3,4,5,6,7,8,9,10,11,12,13,14,15);
}
// B fragment (32x16 f16) from K-contiguous (transposed) storage: lane holds
// column N=lane&15; halves = 16 contiguous K starting at hsel*16.
__device__ __forceinline__ v16h frag_b(const _Float16* p) {
    v8h lo = *(const v8h*)p;
    v8h hi = *(const v8h*)(p + 8);
    return __builtin_shufflevector(lo, hi, 0,1,2,3,4,5,6,7,8,9,10,11,12,13,14,15);
}

// Geometry
#define SEQ   72     // tokens
#define SEQP  80     // padded to 5 WMMA row tiles
#define DIM   256
#define XS    264    // f16/f32 row stride (halves / floats), 528B = 33*16B
#define NHEAD 8
#define HDIM  32
#define VTS   104    // v^T row stride (96 padded K + 8)
#define SCS   88     // score row stride (f32)
#define PSTR  104    // probability row stride (f16)
#define WS    264    // staged-weight row stride (halves)
#define TOPM  20
#define NEGBIG (-3.4e38f)

__global__ __launch_bounds__(256, 1)
void topm_head_kernel(const float* __restrict__ feat,   // (B, D, S)
                      const float* __restrict__ pos,    // (1, S, D)
                      const float* __restrict__ qkv_w,  // (2, D, 3D)
                      const float* __restrict__ qkv_b,  // (2, 3D)
                      const float* __restrict__ proj_w, // (2, D, D)
                      const float* __restrict__ proj_b, // (2, D)
                      const float* __restrict__ ln_g,   // (2, D)
                      const float* __restrict__ ln_b,   // (2, D)
                      const float* __restrict__ w1,     // (D, 128)
                      const float* __restrict__ b1,     // (128)
                      const float* __restrict__ w2,     // (128, 1)
                      const float* __restrict__ b2,     // (1)
                      float* __restrict__ out)          // (B,)
{
    __shared__ float    sx [SEQ  * XS];   // fp32 master activations
    __shared__ _Float16 sxh[SEQP * XS];   // f16 x (QKV A-operand) / attn-out (proj A-operand)
    __shared__ _Float16 sq [SEQP * XS];   // q  (seq x 256)
    __shared__ _Float16 sk [SEQP * XS];   // k  (seq x 256)
    __shared__ _Float16 svt[DIM  * VTS];  // v^T (hd x seqP)
    __shared__ union {
        _Float16 w[128 * WS];             // staged weight block (N x K) / feat staging
        struct {
            float    sc[SEQP * SCS];      // per-head attention scores (f32)
            _Float16 p [SEQP * PSTR];     // per-head softmax probs (f16, K-padded)
        } at;
    } su;
    __shared__ float sred[400];

    const int tid  = threadIdx.x;
    const int lane = tid & 31;
    const int wv   = tid >> 5;            // 8 waves
    const int l15  = lane & 15;
    const int hsel = lane >> 4;
    const int b    = blockIdx.x;

    // ---- init: x = feat^T + pos ------------------------------------------
#ifdef USE_ASYNC_FEAT
    {
        // Async DMA 128 feature rows (36,864B) at a time into the staging
        // region, then transpose + add pos from LDS. Tracked with ASYNCcnt.
        float* stg = (float*)su.w;
        for (int pass = 0; pass < 2; ++pass) {
            const char* src = (const char*)(feat + (size_t)b * DIM * SEQ + (size_t)pass * 128 * SEQ);
            char* dst = (char*)stg;
            for (int i = tid; i < 2304; i += 256) {            // 2304 x 16B
                __builtin_amdgcn_global_load_async_to_lds_b128(
                    AS1_V4I(src + (size_t)i * 16),
                    AS3_V4I(dst + (size_t)i * 16),
                    0, 0);
            }
            __builtin_amdgcn_s_wait_asynccnt(0);
            __syncthreads();
            for (int i = tid; i < 128 * SEQ; i += 256) {
                int d = i / SEQ, s = i % SEQ;
                int c = pass * 128 + d;
                sx[s * XS + c] = stg[i] + pos[s * DIM + c];
            }
            __syncthreads();
        }
    }
#else
    for (int i = tid; i < SEQ * DIM; i += 256) {
        int c = i / SEQ, m = i % SEQ;     // coalesced over feat's fast axis (S)
        sx[m * XS + c] = feat[(size_t)b * DIM * SEQ + (size_t)c * SEQ + m] + pos[m * DIM + c];
    }
#endif
    for (int i = tid; i < (SEQP - SEQ) * XS; i += 256)   // xh rows 72..79 = 0
        sxh[SEQ * XS + i] = (_Float16)0.f;
    for (int i = tid; i < DIM * 2; i += 256) {           // v^T seq cols 80..95 = 0
        v8h z = {};
        *(v8h*)&svt[(i >> 1) * VTS + 80 + (i & 1) * 8] = z;
    }
    __syncthreads();
    for (int i = tid; i < SEQ * DIM; i += 256) {
        int m = i >> 8, c = i & 255;
        sxh[m * XS + c] = (_Float16)sx[m * XS + c];
    }
    __syncthreads();

    const float scale = 0.17677669529663687f;  // 1/sqrt(32)
    const v8f zacc = {};

    for (int l = 0; l < 2; ++l) {
        // ================= QKV GEMM: (72x256) @ (256x768) ==================
        const float* Wq = qkv_w + (size_t)l * DIM * 768;
        const float* Bq = qkv_b + l * 768;
        for (int g = 0; g < 6; ++g) {
            const int n0g = g * 128;
            for (int i = tid; i < 128 * DIM; i += 256) {     // stage W transposed
                int c = i & 127, kk = i >> 7;                // coalesced over N
                su.w[c * WS + kk] = (_Float16)Wq[(size_t)kk * 768 + n0g + c];
                if (g < 5 && (c & 31) == 0)                  // prefetch next group
                    __builtin_prefetch(&Wq[(size_t)kk * 768 + n0g + 128 + c], 0, 1);
            }
            __syncthreads();
            const int n0  = n0g + wv * 16;                   // wave-uniform
            const int n0u = __builtin_amdgcn_readfirstlane(n0);
            // B fragments resident for the whole m-tile loop
            v16h bfr[8];
#pragma unroll
            for (int kt = 0; kt < 8; ++kt)
                bfr[kt] = frag_b(&su.w[(wv * 16 + l15) * WS + kt * 32 + hsel * 16]);
            const float bias = Bq[n0 + l15];
            const _Float16* abase0 = &sxh[l15 * XS + hsel * 16];
            // ---- half-tile software pipeline over 10 half-steps ----------
            // step s: tile mt=s/2, K-half s&1 (4 fragments). Loads for step
            // s+1 are issued before the WMMAs of step s; sched_barrier keeps
            // the groups apart so the wait becomes partial (dscnt<=8).
            v16h hb[2][4];
#pragma unroll
            for (int k = 0; k < 4; ++k) hb[0][k] = frag_a(abase0 + k * 32);
            v8f acc = zacc;
#pragma unroll
            for (int s = 0; s < 10; ++s) {
                const int mt = s >> 1, half = s & 1;
                if (s < 9) {
                    const int ns = s + 1;
                    const _Float16* nb = abase0 + (ns >> 1) * 16 * XS + (ns & 1) * 128;
#pragma unroll
                    for (int k = 0; k < 4; ++k) hb[ns & 1][k] = frag_a(nb + k * 32);
                }
                SCHED_FENCE();
#pragma unroll
                for (int k = 0; k < 4; ++k)
                    acc = WMMA_F16(hb[s & 1][k], bfr[half * 4 + k], acc);
                SCHED_FENCE();
                if (half == 1) {
                    if (n0u < 256) {
#pragma unroll
                        for (int r = 0; r < 8; ++r)
                            sq[(mt * 16 + r + hsel * 8) * XS + n0 + l15] = (_Float16)(acc[r] + bias);
                    } else if (n0u < 512) {
#pragma unroll
                        for (int r = 0; r < 8; ++r)
                            sk[(mt * 16 + r + hsel * 8) * XS + (n0 - 256) + l15] = (_Float16)(acc[r] + bias);
                    } else {
                        // v stored transposed: seq contiguous -> one b128 store
                        v8h pk;
#pragma unroll
                        for (int r = 0; r < 8; ++r) pk[r] = (_Float16)(acc[r] + bias);
                        *(v8h*)&svt[(n0 - 512 + l15) * VTS + mt * 16 + hsel * 8] = pk;
                    }
                    acc = zacc;
                }
            }
            __syncthreads();
        }

        // ================= per-head top-M attention ========================
        for (int h = 0; h < NHEAD; ++h) {
            // scores = q_h @ k_h^T * scale  (5x5 tiles, K=32 exact)
            for (int t = wv; t < 25; t += 8) {
                const int mt = t / 5, nt = t % 5;
                v16h a  = frag_a(&sq[(mt * 16 + l15) * XS + h * 32 + hsel * 16]);
                v16h bf = frag_b(&sk[(nt * 16 + l15) * XS + h * 32 + hsel * 16]);
                v8f acc = WMMA_F16(a, bf, zacc);
#pragma unroll
                for (int r = 0; r < 8; ++r)
                    su.at.sc[(mt * 16 + r + hsel * 8) * SCS + nt * 16 + l15] = acc[r] * scale;
            }
            __syncthreads();
            // top-20 threshold + masked softmax, one thread per row
            if (tid < SEQ) {
                const int r = tid;
                float tk[TOPM];
#pragma unroll
                for (int i = 0; i < TOPM; ++i) tk[i] = NEGBIG;
                for (int j = 0; j < SEQ; ++j) {
                    float v = su.at.sc[r * SCS + j];
                    if (v > tk[TOPM - 1]) {
                        tk[TOPM - 1] = v;
#pragma unroll
                        for (int i = TOPM - 1; i > 0; --i) {
                            if (tk[i] > tk[i - 1]) { float tmp = tk[i - 1]; tk[i - 1] = tk[i]; tk[i] = tmp; }
                            else break;
                        }
                    }
                }
                const float thr = tk[TOPM - 1];
                const float mx  = tk[0];
                float ssum = 0.f;
                for (int j = 0; j < SEQ; ++j) {
                    float v = su.at.sc[r * SCS + j];
                    if (v >= thr) ssum += expf(v - mx);
                }
                const float inv = 1.f / ssum;
                for (int jb = 0; jb < 96; jb += 8) {    // vectorized probability row
                    v8h chunk;
#pragma unroll
                    for (int e = 0; e < 8; ++e) {
                        int j = jb + e;
                        float v = (j < SEQ) ? su.at.sc[r * SCS + j] : NEGBIG;
                        chunk[e] = (v >= thr) ? (_Float16)(expf(v - mx) * inv) : (_Float16)0.f;
                    }
                    *(v8h*)&su.at.p[r * PSTR + jb] = chunk;
                }
            } else if (tid < SEQP) {             // zero padded rows 72..79
                v8h z = {};
                for (int jb = 0; jb < 96; jb += 8) *(v8h*)&su.at.p[tid * PSTR + jb] = z;
            }
            __syncthreads();
            // out_h = p @ v_h  (5x2 tiles, K = 96 padded)
            for (int t = wv; t < 10; t += 8) {
                const int mt = t >> 1, nt = t & 1;
                v16h afr[3], bfr3[3];
#pragma unroll
                for (int kt = 0; kt < 3; ++kt) {
                    afr[kt]  = frag_a(&su.at.p[(mt * 16 + l15) * PSTR + kt * 32 + hsel * 16]);
                    bfr3[kt] = frag_b(&svt[(h * 32 + nt * 16 + l15) * VTS + kt * 32 + hsel * 16]);
                }
                v8f acc = zacc;
#pragma unroll
                for (int kt = 0; kt < 3; ++kt)
                    acc = WMMA_F16(afr[kt], bfr3[kt], acc);
#pragma unroll
                for (int r = 0; r < 8; ++r)
                    sxh[(mt * 16 + r + hsel * 8) * XS + h * 32 + nt * 16 + l15] = (_Float16)acc[r];
            }
            __syncthreads();
        }

        // ================= proj GEMM + residual ===========================
        const float* Wp = proj_w + (size_t)l * DIM * DIM;
        const float* Bp = proj_b + l * DIM;
        for (int g = 0; g < 2; ++g) {
            const int n0g = g * 128;
            for (int i = tid; i < 128 * DIM; i += 256) {
                int c = i & 127, kk = i >> 7;
                su.w[c * WS + kk] = (_Float16)Wp[(size_t)kk * DIM + n0g + c];
                if (g < 1 && (c & 31) == 0)
                    __builtin_prefetch(&Wp[(size_t)kk * DIM + n0g + 128 + c], 0, 1);
            }
            __syncthreads();
            const int n0 = n0g + wv * 16;
            v16h bfr[8];
#pragma unroll
            for (int kt = 0; kt < 8; ++kt)
                bfr[kt] = frag_b(&su.w[(wv * 16 + l15) * WS + kt * 32 + hsel * 16]);
            const float bias = Bp[n0 + l15];
            const _Float16* abase0 = &sxh[l15 * XS + hsel * 16];
            v16h hb[2][4];
#pragma unroll
            for (int k = 0; k < 4; ++k) hb[0][k] = frag_a(abase0 + k * 32);
            v8f acc = zacc;
#pragma unroll
            for (int s = 0; s < 10; ++s) {
                const int mt = s >> 1, half = s & 1;
                if (s < 9) {
                    const int ns = s + 1;
                    const _Float16* nb = abase0 + (ns >> 1) * 16 * XS + (ns & 1) * 128;
#pragma unroll
                    for (int k = 0; k < 4; ++k) hb[ns & 1][k] = frag_a(nb + k * 32);
                }
                SCHED_FENCE();
#pragma unroll
                for (int k = 0; k < 4; ++k)
                    acc = WMMA_F16(hb[s & 1][k], bfr[half * 4 + k], acc);
                SCHED_FENCE();
                if (half == 1) {
#pragma unroll
                    for (int r = 0; r < 8; ++r) {
                        const int row = mt * 16 + r + hsel * 8;
                        if (row < SEQ) sx[row * XS + n0 + l15] += acc[r] + bias;  // residual
                    }
                    acc = zacc;
                }
            }
            __syncthreads();
        }

        // ================= LayerNorm (fp32), refresh f16 copy =============
        const float* G  = ln_g + l * DIM;
        const float* Bb = ln_b + l * DIM;
        for (int r = wv; r < SEQ; r += 8) {
            float s = 0.f;
            for (int c = lane; c < DIM; c += 32) s += sx[r * XS + c];
#pragma unroll
            for (int m = 16; m >= 1; m >>= 1) s += __shfl_xor(s, m, 32);
            const float mu = s * (1.f / 256.f);
            float v = 0.f;
            for (int c = lane; c < DIM; c += 32) {
                float d = sx[r * XS + c] - mu; v += d * d;
            }
#pragma unroll
            for (int m = 16; m >= 1; m >>= 1) v += __shfl_xor(v, m, 32);
            const float rs = rsqrtf(v * (1.f / 256.f) + 1e-5f);
            for (int c = lane; c < DIM; c += 32) {
                float val = (sx[r * XS + c] - mu) * rs * G[c] + Bb[c];
                sx[r * XS + c]  = val;
                sxh[r * XS + c] = (_Float16)val;
            }
        }
        __syncthreads();
    }

    // ================= mean-pool + classifier MLP =========================
    for (int c = tid; c < DIM; c += 256) {
        float s = 0.f;
        for (int m = 0; m < SEQ; ++m) s += sx[m * XS + c];
        sred[c] = s * (1.f / 72.f);
    }
    __syncthreads();
    if (tid < 128) {
        float acc = b1[tid];
        for (int kk = 0; kk < DIM; ++kk) acc += sred[kk] * w1[kk * 128 + tid];
        sred[272 + tid] = acc > 0.f ? acc : 0.f;
    }
    __syncthreads();
    if (wv == 0) {
        float s = 0.f;
        for (int j = lane; j < 128; j += 32) s += sred[272 + j] * w2[j];
#pragma unroll
        for (int m = 16; m >= 1; m >>= 1) s += __shfl_xor(s, m, 32);
        if (lane == 0) out[b] = s + b2[0];
    }
}

extern "C" void kernel_launch(void* const* d_in, const int* in_sizes, int n_in,
                              void* d_out, int out_size, void* d_ws, size_t ws_size,
                              hipStream_t stream) {
    const float* feat   = (const float*)d_in[0];
    const float* pos    = (const float*)d_in[1];
    const float* qkv_w  = (const float*)d_in[2];
    const float* qkv_b  = (const float*)d_in[3];
    const float* proj_w = (const float*)d_in[4];
    const float* proj_b = (const float*)d_in[5];
    const float* ln_g   = (const float*)d_in[6];
    const float* ln_b   = (const float*)d_in[7];
    const float* w1     = (const float*)d_in[8];
    const float* b1     = (const float*)d_in[9];
    const float* w2     = (const float*)d_in[10];
    const float* b2     = (const float*)d_in[11];
    float* out = (float*)d_out;

    const int B = in_sizes[0] / (DIM * SEQ);   // 1920 sequences
    topm_head_kernel<<<dim3(B), dim3(256), 0, stream>>>(
        feat, pos, qkv_w, qkv_b, proj_w, proj_b, ln_g, ln_b, w1, b1, w2, b2, out);
}
